// LSTM_CNN_CRF_43198781063231
// MI455X (gfx1250) — matrix operand order repaired and checked by reading, not compiled, not run on hardware
//
#include <hip/hip_runtime.h>
#include <hip/hip_bf16.h>
#include <cstdint>
#include <cstddef>

typedef __attribute__((ext_vector_type(16))) _Float16 v16h;
typedef __attribute__((ext_vector_type(8)))  float    v8f;
typedef unsigned int u32x4 __attribute__((ext_vector_type(4)));
typedef int          i32x4 __attribute__((ext_vector_type(4)));
typedef int          i32x8 __attribute__((ext_vector_type(8)));

// Problem constants
constexpr int B_  = 64;
constexpr int S_  = 512;
constexpr int W_  = 20;
constexpr int DW_ = 300;
constexpr int DC_ = 100;
constexpr int H_  = 256;
constexpr int T_  = 7;
constexpr int K_  = 3;
constexpr int D_  = DW_ + DC_;   // 400
constexpr int DP_ = 416;         // K padded to 13*32 for WMMA
constexpr int KT_IH = 13;        // k-tiles input projection
constexpr int KT_HH = 8;         // k-tiles recurrence (256/32)
constexpr int G4H = 4 * H_;      // 1024 gate width
constexpr int NT_ = G4H / 16;    // 64 n-tiles
constexpr int BS_ = B_ * S_;     // 32768 tokens

__device__ __forceinline__ float sigf(float x) { return 1.0f / (1.0f + __expf(-x)); }

// ---------------------------------------------------------------------------
// TDM: async-load a 16-row x 1024-element f16 tile (row stride S_*G4H elems)
// from global xg into LDS. Descriptor per CDNA5 ISA.
// Issued by one wave; completion tracked with TENSORcnt.
// ---------------------------------------------------------------------------
__device__ __forceinline__ void tdm_load_row_tile(const _Float16* gsrc, unsigned int lds_off) {
  unsigned long long ga = (unsigned long long)(uintptr_t)gsrc;
  u32x4 g0;
  g0[0] = 1u;                                   // count=1 valid user descriptor
  g0[1] = lds_off;                              // lds_addr (bytes)
  g0[2] = (unsigned int)(ga & 0xffffffffull);   // global_addr[95:64]
  g0[3] = (unsigned int)(ga >> 32) | 0x80000000u; // global_addr[120:96] | type=2
  i32x8 g1;
  g1[0] = 1 << 16;              // data_size=1 (2 bytes), wg_mask=0
  g1[1] = (int)(1024u << 16);   // tensor_dim0 = 1024 (bits 63:48)
  g1[2] = (int)(16u << 16);     // tensor_dim1 = 16   (bits 95:80)
  g1[3] = (int)(1024u << 16);   // tile_dim0 = 1024   (bits 127:112)
  g1[4] = 16;                   // tile_dim1 = 16     (bits 143:128)
  g1[5] = S_ * G4H;             // tensor_dim0_stride = 524288 elems (bits 191:160)
  g1[6] = 0;
  g1[7] = 0;
  i32x4 z4; z4[0] = 0; z4[1] = 0; z4[2] = 0; z4[3] = 0;
#if defined(__clang_major__) && (__clang_major__ >= 23)
  i32x8 z8; z8[0]=0; z8[1]=0; z8[2]=0; z8[3]=0; z8[4]=0; z8[5]=0; z8[6]=0; z8[7]=0;
  __builtin_amdgcn_tensor_load_to_lds(g0, g1, z4, z4, z8, 0);
#else
  __builtin_amdgcn_tensor_load_to_lds(g0, g1, z4, z4, 0);
#endif
}

// ---------------------------------------------------------------------------
// Weight packing into WMMA B-fragment layout.
// B matrix (32K x 16N) per 16x16x32 tile: lane l holds column n = l&15,
// k-range [ (l>>4)*16 , +16 ), element e -> k_local = (l>>4)*16 + e.
// ---------------------------------------------------------------------------
__global__ void pack_wih_kernel(const float* __restrict__ wf,
                                const float* __restrict__ wb,
                                _Float16* __restrict__ out) {
  int idx = blockIdx.x * blockDim.x + threadIdx.x;
  int total = 2 * NT_ * KT_IH * 512;
  if (idx >= total) return;
  int e = idx & 15;
  int lane = (idx >> 4) & 31;
  int t = idx >> 9;
  int kt = t % KT_IH; t /= KT_IH;
  int nt = t & 63;
  int dir = t >> 6;
  int n = nt * 16 + (lane & 15);
  int k = kt * 32 + (lane >> 4) * 16 + e;
  const float* w = dir ? wb : wf;
  float v = (k < D_) ? w[(size_t)n * D_ + k] : 0.0f;
  out[idx] = (_Float16)v;
}

__global__ void pack_whh_kernel(const float* __restrict__ wf,
                                const float* __restrict__ wb,
                                _Float16* __restrict__ out) {
  int idx = blockIdx.x * blockDim.x + threadIdx.x;
  int total = 2 * NT_ * KT_HH * 512;
  if (idx >= total) return;
  int e = idx & 15;
  int lane = (idx >> 4) & 31;
  int t = idx >> 9;
  int kt = t & 7; t >>= 3;
  int nt = t & 63;
  int dir = t >> 6;
  int n = nt * 16 + (lane & 15);
  int k = kt * 32 + (lane >> 4) * 16 + e;
  const float* w = dir ? wb : wf;
  out[idx] = (_Float16)w[(size_t)n * H_ + k];
}

__global__ void pack_bias_kernel(const float* __restrict__ bihf, const float* __restrict__ bhhf,
                                 const float* __restrict__ bihb, const float* __restrict__ bhhb,
                                 float* __restrict__ out) {
  int idx = blockIdx.x * blockDim.x + threadIdx.x;
  if (idx >= 2 * G4H) return;
  int dir = idx >> 10;
  int n = idx & 1023;
  out[idx] = dir ? (bihb[n] + bhhb[n]) : (bihf[n] + bhhf[n]);
}

// ---------------------------------------------------------------------------
// Char CNN (embed -> conv1d K=3 -> relu -> max) + packed f16 input rows
// x = [word_emb | char_feat | zero pad], length DP_=416. One block per token.
// ---------------------------------------------------------------------------
__global__ void __launch_bounds__(128) charcnn_kernel(const float* __restrict__ word,
                                                      const int* __restrict__ chars,
                                                      const float* __restrict__ cemb,
                                                      const float* __restrict__ cw,
                                                      const float* __restrict__ cb,
                                                      _Float16* __restrict__ xh) {
  __shared__ float e_sm[W_ * DC_];
  __shared__ float feat[DC_];
  int tok = blockIdx.x;
  int tid = threadIdx.x;
  for (int idx = tid; idx < W_ * DC_; idx += 128) {
    int w = idx / DC_;
    int c = idx - w * DC_;
    int id = chars[tok * W_ + w];
    e_sm[idx] = cemb[(size_t)id * DC_ + c];
  }
  __syncthreads();
  if (tid < DC_) {
    float acc[W_ - K_ + 1];
#pragma unroll
    for (int p = 0; p < W_ - K_ + 1; p++) acc[p] = 0.0f;
    for (int c = 0; c < DC_; c++) {
      float ecol[W_];
#pragma unroll
      for (int w = 0; w < W_; w++) ecol[w] = e_sm[w * DC_ + c];
#pragma unroll
      for (int k = 0; k < K_; k++) {
        float wv = cw[((size_t)tid * DC_ + c) * K_ + k];
#pragma unroll
        for (int p = 0; p < W_ - K_ + 1; p++) acc[p] += ecol[p + k] * wv;
      }
    }
    float b = cb[tid];
    float mx = 0.0f;  // relu floor
#pragma unroll
    for (int p = 0; p < W_ - K_ + 1; p++) {
      float y = acc[p] + b;
      y = y > 0.0f ? y : 0.0f;
      mx = y > mx ? y : mx;
    }
    feat[tid] = mx;
  }
  __syncthreads();
  for (int idx = tid; idx < DP_; idx += 128) {
    float v;
    if (idx < DW_) v = word[(size_t)tok * DW_ + idx];
    else if (idx < D_) v = feat[idx - DW_];
    else v = 0.0f;
    xh[(size_t)tok * DP_ + idx] = (_Float16)v;
  }
}

// ---------------------------------------------------------------------------
// Input projection GEMM: xg[dir][m][n] = x[m].w_ih[dir][n] + b_ih + b_hh.
// M=32768 (tiles of 16), N=1024, K=416. Output stored f16 (consumed as WMMA-C
// initializer via LDS next kernel -> halves HBM traffic to 134 MB).
// ---------------------------------------------------------------------------
__global__ void __launch_bounds__(256) xgemm_kernel(const _Float16* __restrict__ xh,
                                                    const _Float16* __restrict__ wih,
                                                    const float* __restrict__ biast,
                                                    _Float16* __restrict__ xg) {
  __shared__ __align__(32) _Float16 a_sm[KT_IH * 512];
  int dir = blockIdx.x & 1;
  int mt = blockIdx.x >> 1;
  int tid = threadIdx.x;
  for (int idx = tid; idx < KT_IH * 512; idx += 256) {
    int kt = idx >> 9;
    int r = idx & 511;
    int lane = r >> 4;
    int e = r & 15;
    int m = mt * 16 + (lane & 15);
    int kl = ((lane >> 4) << 4) | e;
    int k = kt * 32 + kl;
    a_sm[idx] = xh[(size_t)m * DP_ + k];
  }
  __syncthreads();
  int wv = tid >> 5, lane = tid & 31, ln = lane & 15, lh = lane >> 4;
  for (int i = 0; i < 8; i++) {
    int nt = wv * 8 + i;
    int n = nt * 16 + ln;
    float bias = biast[dir * G4H + n];
    v8f acc;
#pragma unroll
    for (int r = 0; r < 8; r++) acc[r] = bias;
    const _Float16* bptr = wih + ((size_t)(dir * NT_ + nt) * KT_IH) * 512 + lane * 16;
#pragma unroll
    for (int kt = 0; kt < KT_IH; kt++) {
      v16h a = *reinterpret_cast<const v16h*>(&a_sm[kt * 512 + lane * 16]);
      v16h bb = *reinterpret_cast<const v16h*>(bptr + (size_t)kt * 512);
      acc = __builtin_amdgcn_wmma_f32_16x16x32_f16(false, a, false, bb, (short)0, acc, false, false);
    }
#pragma unroll
    for (int r = 0; r < 8; r++) {
      int m = mt * 16 + r + 8 * lh;
      xg[((size_t)dir * BS_ + m) * G4H + n] = (_Float16)acc[r];
    }
  }
}

// ---------------------------------------------------------------------------
// Persistent BiLSTM recurrence. 8 blocks = 2 dirs x 4 batch-slices (16 rows).
// TDM double-buffers the next step's 16x1024 f16 xg tile into LDS while all
// 8 waves run the 64x8 WMMA recurrent GEMM of the current step. Gates are
// written in place over the consumed xg tile. Thread j owns hidden unit j
// for all 16 rows; (h,c) live in registers; h re-packed into the LDS
// A-fragment layout for the next step's WMMA.
// ---------------------------------------------------------------------------
__global__ void __launch_bounds__(256) lstm_kernel(const _Float16* __restrict__ xg,
                                                   const _Float16* __restrict__ whh,
                                                   const int* __restrict__ lengths,
                                                   float* __restrict__ hout) {
  __shared__ __align__(32) _Float16 h_frag[KT_HH * 512];
  __shared__ __align__(16) _Float16 xbuf[2][16 * G4H];   // xg tile in, gates out
  __shared__ int len_sm[16];
  int dir = blockIdx.x >> 2;
  int b0 = (blockIdx.x & 3) * 16;
  int tid = threadIdx.x;
  int wv = tid >> 5, lane = tid & 31, ln = lane & 15, lh = lane >> 4;
  if (tid < 16) len_sm[tid] = lengths[b0 + tid];
  for (int idx = tid; idx < KT_HH * 512; idx += 256) h_frag[idx] = (_Float16)0.0f;
  float cst[16], hst[16];
#pragma unroll
  for (int r = 0; r < 16; r++) { cst[r] = 0.0f; hst[r] = 0.0f; }
  int j = tid;  // hidden unit owned by this thread

  const _Float16* xg0 = xg + (((size_t)dir * B_ + b0) * S_) * G4H;  // + t*G4H per step
  int t0 = dir ? (S_ - 1) : 0;
  if (tid < 32) tdm_load_row_tile(xg0 + (size_t)t0 * G4H, (unsigned int)(uintptr_t)&xbuf[0][0]);

  int cur = 0;
  for (int step = 0; step < S_; ++step) {
    int t = dir ? (S_ - 1 - step) : step;
    // ---- prefetch next tile via TDM, wait for current ----
    if (tid < 32) {
      if (step + 1 < S_) {
        int tn = dir ? (t - 1) : (t + 1);
        tdm_load_row_tile(xg0 + (size_t)tn * G4H, (unsigned int)(uintptr_t)&xbuf[cur ^ 1][0]);
        __builtin_amdgcn_s_wait_tensorcnt((short)1);
      } else {
        __builtin_amdgcn_s_wait_tensorcnt((short)0);
      }
    }
    __syncthreads();
    // ---- GEMM phase: gates = xg(t) + h @ w_hhT ----
    for (int i = 0; i < 8; i++) {
      int nt = wv * 8 + i;
      int n = nt * 16 + ln;
      v8f acc;
#pragma unroll
      for (int r = 0; r < 8; r++) {
        int brow = r + 8 * lh;
        acc[r] = (float)xbuf[cur][brow * G4H + n];
      }
      const _Float16* bptr = whh + ((size_t)(dir * NT_ + nt) * KT_HH) * 512 + lane * 16;
      if (i < 7)
        __builtin_prefetch((const void*)(whh + ((size_t)(dir * NT_ + nt + 1) * KT_HH) * 512 + lane * 16), 0, 0);
#pragma unroll
      for (int kt = 0; kt < KT_HH; kt++) {
        v16h a = *reinterpret_cast<const v16h*>(&h_frag[kt * 512 + lane * 16]);
        v16h bb = *reinterpret_cast<const v16h*>(bptr + (size_t)kt * 512);
        acc = __builtin_amdgcn_wmma_f32_16x16x32_f16(false, a, false, bb, (short)0, acc, false, false);
      }
#pragma unroll
      for (int r = 0; r < 8; r++) {
        int brow = r + 8 * lh;
        xbuf[cur][brow * G4H + n] = (_Float16)acc[r];  // gates in place
      }
    }
    __syncthreads();
    // ---- elementwise gate phase ----
#pragma unroll
    for (int brow = 0; brow < 16; ++brow) {
      float gi = (float)xbuf[cur][brow * G4H + j];
      float gf = (float)xbuf[cur][brow * G4H + j + H_];
      float gg = (float)xbuf[cur][brow * G4H + j + 2 * H_];
      float go = (float)xbuf[cur][brow * G4H + j + 3 * H_];
      float c_new = sigf(gf) * cst[brow] + sigf(gi) * tanhf(gg);
      float h_new = sigf(go) * tanhf(c_new);
      bool msk = t < len_sm[brow];
      float h2 = msk ? h_new : hst[brow];
      float c2 = msk ? c_new : cst[brow];
      cst[brow] = c2;
      hst[brow] = h2;
      int b = b0 + brow;
      hout[(((size_t)dir * B_ + b) * S_ + t) * H_ + j] = h2;
      // write back into A-fragment layout for next step
      int kl = j & 31;
      int lhh = kl >> 4;
      int e = kl & 15;
      int lanew = brow + (lhh << 4);
      h_frag[(j >> 5) * 512 + lanew * 16 + e] = (_Float16)h2;
    }
    __syncthreads();
    cur ^= 1;
  }
}

// ---------------------------------------------------------------------------
// Emissions: em[tok][t] = concat(hf,hb) . lin_w[t] + lin_b[t], stride-8 pad.
// ---------------------------------------------------------------------------
__global__ void __launch_bounds__(256) emis_kernel(const float* __restrict__ hout,
                                                   const float* __restrict__ lw,
                                                   const float* __restrict__ lb,
                                                   float* __restrict__ em) {
  int g = blockIdx.x * blockDim.x + threadIdx.x;
  int tok = g >> 3;
  int tt = g & 7;
  if (tok >= BS_) return;
  if (tt < T_) {
    const float* hf = hout + (size_t)tok * H_;
    const float* hb = hout + ((size_t)BS_ + tok) * H_;
    const float* w = lw + (size_t)tt * (2 * H_);
    float acc = lb[tt];
    for (int k = 0; k < H_; k++) acc += hf[k] * w[k];
    for (int k = 0; k < H_; k++) acc += hb[k] * w[H_ + k];
    em[(size_t)tok * 8 + tt] = acc;
  } else {
    em[(size_t)tok * 8 + tt] = 0.0f;
  }
}

// ---------------------------------------------------------------------------
// CRF forward (log-norm) + gold score; one wave32 per batch element.
// ---------------------------------------------------------------------------
__global__ void __launch_bounds__(256) crf_fwd_kernel(const float* __restrict__ em,
                                                      const int* __restrict__ tags,
                                                      const int* __restrict__ lengths,
                                                      const float* __restrict__ startv,
                                                      const float* __restrict__ endv,
                                                      const float* __restrict__ trans,
                                                      float* __restrict__ nd) {
  int b = (blockIdx.x * blockDim.x + threadIdx.x) >> 5;
  int lane = threadIdx.x & 31;
  if (b >= B_) return;
  int len = lengths[b];
  int j = lane < T_ ? lane : 0;
  const float* emb = em + (size_t)b * S_ * 8;
  float alpha = (lane < T_) ? (startv[j] + emb[j]) : -1e30f;
  float tr[T_];
#pragma unroll
  for (int i = 0; i < T_; i++) tr[i] = trans[i * T_ + j];
  for (int s = 1; s < S_; s++) {
    float m = -1e30f;
    float cand[T_];
#pragma unroll
    for (int i = 0; i < T_; i++) {
      float ai = __shfl(alpha, i, 32);
      cand[i] = ai + tr[i];
      m = fmaxf(m, cand[i]);
    }
    float sum = 0.0f;
#pragma unroll
    for (int i = 0; i < T_; i++) sum += __expf(cand[i] - m);
    float nxt = m + __logf(sum) + emb[s * 8 + j];
    if (lane < T_ && s < len) alpha = nxt;
  }
  float val = (lane < T_) ? (alpha + endv[j]) : -1e30f;
  float mm = -1e30f;
#pragma unroll
  for (int i = 0; i < T_; i++) mm = fmaxf(mm, __shfl(val, i, 32));
  float ss = 0.0f;
#pragma unroll
  for (int i = 0; i < T_; i++) ss += __expf(__shfl(val, i, 32) - mm);
  float den = mm + __logf(ss);
  if (lane == 0) {
    const int* tg = tags + b * S_;
    int prev = tg[0];
    float score = startv[prev] + emb[prev];
    for (int s = 1; s < S_; s++) {
      if (s < len) {
        int cur = tg[s];
        score += trans[prev * T_ + cur] + emb[s * 8 + cur];
        prev = cur;
      }
    }
    score += endv[prev];
    nd[b] = score - den;
  }
}

__global__ void loss_kernel(const float* __restrict__ nd, float* __restrict__ out) {
  __shared__ float red[64];
  int tid = threadIdx.x;
  red[tid] = nd[tid];
  __syncthreads();
  for (int off = 32; off > 0; off >>= 1) {
    if (tid < off) red[tid] += red[tid + off];
    __syncthreads();
  }
  if (tid == 0) out[0] = -red[0] / (float)B_;
}

// ---------------------------------------------------------------------------
// Viterbi: one wave32 per batch element; backpointers in ws; lane 0 backtrace.
// ---------------------------------------------------------------------------
__global__ void __launch_bounds__(256) viterbi_kernel(const float* __restrict__ em,
                                                      const int* __restrict__ lengths,
                                                      const float* __restrict__ startv,
                                                      const float* __restrict__ endv,
                                                      const float* __restrict__ trans,
                                                      unsigned char* __restrict__ bp,
                                                      float* __restrict__ out) {
  int b = (blockIdx.x * blockDim.x + threadIdx.x) >> 5;
  int lane = threadIdx.x & 31;
  if (b >= B_) return;
  int len = lengths[b];
  int j = lane < T_ ? lane : 0;
  const float* emb = em + (size_t)b * S_ * 8;
  float score = (lane < T_) ? (startv[j] + emb[j]) : -1e30f;
  float tr[T_];
#pragma unroll
  for (int i = 0; i < T_; i++) tr[i] = trans[i * T_ + j];
  unsigned char* bpb = bp + (size_t)b * S_ * 8;
  for (int s = 1; s < S_; s++) {
    float best = -1e30f;
    int arg = 0;
#pragma unroll
    for (int i = 0; i < T_; i++) {
      float v = __shfl(score, i, 32) + tr[i];
      if (v > best) { best = v; arg = i; }
    }
    if (lane < T_) bpb[s * 8 + j] = (unsigned char)arg;
    float nxt = best + emb[s * 8 + j];
    if (lane < T_ && s < len) score = nxt;
  }
  float sc_end = (lane < T_) ? (score + endv[j]) : -1e30f;
  float best = -1e30f;
  int last = 0;
#pragma unroll
  for (int i = 0; i < T_; i++) {
    float v = __shfl(sc_end, i, 32);
    if (v > best) { best = v; last = i; }
  }
  if (lane == 0) {
    int cur = last;
    for (int s = S_ - 1; s >= 1; --s) {
      out[1 + b * S_ + s] = (s < len) ? (float)cur : 0.0f;
      if (s < len) cur = bpb[s * 8 + cur];
    }
    out[1 + b * S_] = (float)cur;  // mask[b][0] always true (len >= S/2)
  }
}

// ---------------------------------------------------------------------------
extern "C" void kernel_launch(void* const* d_in, const int* in_sizes, int n_in,
                              void* d_out, int out_size, void* d_ws, size_t ws_size,
                              hipStream_t stream) {
  (void)in_sizes; (void)n_in; (void)out_size; (void)ws_size;
  const float* word  = (const float*)d_in[0];
  const int* chars   = (const int*)d_in[1];
  const int* tags    = (const int*)d_in[2];
  const int* lengths = (const int*)d_in[3];
  // d_in[4] token_mask is recomputed from lengths (avoids bool dtype ambiguity)
  const float* cemb = (const float*)d_in[5];
  const float* cw   = (const float*)d_in[6];
  const float* cb   = (const float*)d_in[7];
  const float* wihf = (const float*)d_in[8];
  const float* whhf = (const float*)d_in[9];
  const float* bihf = (const float*)d_in[10];
  const float* bhhf = (const float*)d_in[11];
  const float* wihb = (const float*)d_in[12];
  const float* whhb = (const float*)d_in[13];
  const float* bihb = (const float*)d_in[14];
  const float* bhhb = (const float*)d_in[15];
  const float* lw   = (const float*)d_in[16];
  const float* lb   = (const float*)d_in[17];
  const float* cst  = (const float*)d_in[18];
  const float* cen  = (const float*)d_in[19];
  const float* ctr  = (const float*)d_in[20];

  char* ws = (char*)d_ws;
  size_t off = 0;
  auto alloc = [&](size_t bytes) -> void* {
    void* p = ws + off;
    off = (off + bytes + 255) & ~(size_t)255;
    return p;
  };
  _Float16* xh     = (_Float16*)alloc((size_t)BS_ * DP_ * 2);
  _Float16* wih_p  = (_Float16*)alloc((size_t)2 * NT_ * KT_IH * 512 * 2);
  _Float16* whh_p  = (_Float16*)alloc((size_t)2 * NT_ * KT_HH * 512 * 2);
  float* biast     = (float*)alloc((size_t)2 * G4H * 4);
  _Float16* xg     = (_Float16*)alloc((size_t)2 * BS_ * G4H * 2);
  float* hout      = (float*)alloc((size_t)2 * BS_ * H_ * 4);
  float* em        = (float*)alloc((size_t)BS_ * 8 * 4);
  float* nd        = (float*)alloc(64 * 4);
  unsigned char* bp = (unsigned char*)alloc((size_t)BS_ * 8);
  float* out = (float*)d_out;

  pack_wih_kernel<<<(2 * NT_ * KT_IH * 512 + 255) / 256, 256, 0, stream>>>(wihf, wihb, wih_p);
  pack_whh_kernel<<<(2 * NT_ * KT_HH * 512 + 255) / 256, 256, 0, stream>>>(whhf, whhb, whh_p);
  pack_bias_kernel<<<(2 * G4H + 255) / 256, 256, 0, stream>>>(bihf, bhhf, bihb, bhhb, biast);
  charcnn_kernel<<<BS_, 128, 0, stream>>>(word, chars, cemb, cw, cb, xh);
  xgemm_kernel<<<(BS_ / 16) * 2, 256, 0, stream>>>(xh, wih_p, biast, xg);
  lstm_kernel<<<8, 256, 0, stream>>>(xg, whh_p, lengths, hout);
  emis_kernel<<<(BS_ * 8) / 256, 256, 0, stream>>>(hout, lw, lb, em);
  crf_fwd_kernel<<<8, 256, 0, stream>>>(em, tags, lengths, cst, cen, ctr, nd);
  loss_kernel<<<1, 64, 0, stream>>>(nd, out);
  viterbi_kernel<<<8, 256, 0, stream>>>(em, lengths, cst, cen, ctr, bp, out);
}
